// MultiheadLeap_44229573214608
// MI455X (gfx1250) — matrix operand order repaired and checked by reading, not compiled, not run on hardware
//
#include <hip/hip_runtime.h>
#include <hip/hip_bf16.h>

typedef __attribute__((ext_vector_type(16))) __bf16 v16bf;
typedef __attribute__((ext_vector_type(8)))  __bf16 v8bf;
typedef __attribute__((ext_vector_type(8)))  float  v8f;

#define NHEAD   16
#define HD      64
#define HID     (NHEAD * HD)      // 1024
#define SEQ     8192
#define TILE    128               // == WINDOW
#define SPAN    256               // prev tile + current tile
#define LSTR    264               // padded LDS row stride (bf16 elems), 16B aligned
#define SCALE_F 0.125f            // 1/sqrt(64)
#define EPS_F   1e-5f

__global__ __launch_bounds__(256)
void leap_wmma_kernel(const float* __restrict__ q, const float* __restrict__ f,
                      const float* __restrict__ kk, const float* __restrict__ vv,
                      float* __restrict__ out)
{
    // Transposed bf16 staging: [channel][span]  (padded rows -> conflict-light b128 reads)
    __shared__ __align__(16) __bf16 s_wkT[HD * LSTR];   // 33.0 KB
    __shared__ __align__(16) __bf16 s_wvT[HD * LSTR];   // 33.0 KB
    __shared__ float s_w[SPAN];                         // focus weights (logit accum first)
    __shared__ float s_invW[TILE];                      // 1/(window_sum(w)+eps)

    const int tid  = threadIdx.x;
    const int lane = tid & 31;

    const int tile  = blockIdx.x;   // 0..63
    const int head  = blockIdx.y;   // 0..15
    const int batch = blockIdx.z;   // 0..B-1

    const long baseBH = (long)batch * SEQ * HID + (long)head * HD;
    const int  t0     = tile * TILE;           // first output seq position
    // local span index sl in [0,256) -> global seq gs = t0 - 128 + sl

    if (tid < SPAN) s_w[tid] = 0.f;
    __syncthreads();

    // ---- Phase A: focus logits  w_logit[s] = sum_c f*k ------------------
    // 4 span-positions per pass; each position reduced by a wave-pair.
    for (int it = 0; it < SPAN / 4; ++it) {
        const int sl = it * 4 + (tid >> 6);
        const int c  = tid & 63;
        const int gs = t0 - TILE + sl;
        float p = 0.f;
        if (gs >= 0) {
            const long idx = baseBH + (long)gs * HID + c;
            p = f[idx] * kk[idx];
        }
        #pragma unroll
        for (int m = 16; m >= 1; m >>= 1) p += __shfl_xor(p, m, 32);
        if (lane == 0) atomicAdd(&s_w[sl], p);
    }
    __syncthreads();

    // logits -> weights (positions before seq start contribute exactly 0)
    if (tid < SPAN) {
        const int gs = t0 - TILE + tid;
        s_w[tid] = (gs >= 0) ? __expf(s_w[tid] * SCALE_F) : 0.f;
    }
    __syncthreads();

    // ---- Phase B: stage w*k, w*v transposed in bf16 ---------------------
    for (int it = 0; it < SPAN / 4; ++it) {
        const int sl = it * 4 + (tid >> 6);
        const int c  = tid & 63;
        const int gs = t0 - TILE + sl;
        const float w = s_w[sl];
        float kvv = 0.f, vvv = 0.f;
        if (gs >= 0) {
            const long idx = baseBH + (long)gs * HID + c;
            kvv = kk[idx];
            vvv = vv[idx];
        }
        s_wkT[c * LSTR + sl] = (__bf16)(w * kvv);
        s_wvT[c * LSTR + sl] = (__bf16)(w * vvv);
    }

    // ---- Phase C: scalar window normalizers (independent of B) ---------
    if (tid < TILE) {
        float acc = 0.f;
        for (int j = 1; j <= TILE; ++j) acc += s_w[tid + j];   // window [r+1, r+128]
        s_invW[tid] = 1.f / (acc + EPS_F);
        // prefetch this row of q while WMMA phase warms up
        __builtin_prefetch(&q[baseBH + (long)(t0 + tid) * HID], 0, 1);
    }
    __syncthreads();

    // ---- Phase D: banded WMMA  D = Band(128x256) @ X(256x64) -----------
    // wave w owns output row-tile Rt = w (rows 16w..16w+15), all 4 col-tiles,
    // both matrices (8 f32 accumulator tiles). Rt is wave-uniform: pin it to
    // an SGPR so the K-chunk loop is scalar (s_cbranch, no exec masking).
    const int Rt    = __builtin_amdgcn_readfirstlane(tid >> 5);
    const int nlane = lane & 15;
    const int hiK   = lane >> 4;
    const int rlane = Rt * 16 + nlane;          // A-matrix row for this lane

    v8f accK[4], accV[4];
    #pragma unroll
    for (int ct = 0; ct < 4; ++ct)
        #pragma unroll
        for (int i = 0; i < 8; ++i) { accK[ct][i] = 0.f; accV[ct][i] = 0.f; }

    // band rows [16Rt,16Rt+15] touch span cols [16Rt+1, 16Rt+143] -> 5 K-chunks
    const int cBeg = (16 * Rt + 1)   >> 5;
    const int cEnd = (16 * Rt + 143) >> 5;

    for (int ck = cBeg; ck <= cEnd; ++ck) {
        // A fragment: 16x32 bf16, contiguous-K layout (matches B):
        // K = 16*hiK + e
        v16bf a;
        #pragma unroll
        for (int e = 0; e < 16; ++e) {
            const int K = hiK * 16 + e;
            const int s = ck * 32 + K;
            a[e] = (__bf16)((s >= rlane + 1 && s <= rlane + 128) ? 1.0f : 0.0f);
        }

        // Issue ALL 16 ds_load_b128 for this chunk up front, then the 8 WMMAs:
        // lets the backend overlap LDS latency with the WMMA tail instead of
        // draining dscnt before every matrix op.
        const int sbase = ck * 32 + hiK * 16;   // K contiguous per lane-half (B layout)
        v8bf kfrag[4][2], vfrag[4][2];
        #pragma unroll
        for (int ct = 0; ct < 4; ++ct) {
            const int col = ct * 16 + nlane;
            const v8bf* pk = (const v8bf*)&s_wkT[col * LSTR + sbase];
            const v8bf* pv = (const v8bf*)&s_wvT[col * LSTR + sbase];
            kfrag[ct][0] = pk[0]; kfrag[ct][1] = pk[1];
            vfrag[ct][0] = pv[0]; vfrag[ct][1] = pv[1];
        }
        #pragma unroll
        for (int ct = 0; ct < 4; ++ct) {
            v16bf bK, bV;
            #pragma unroll
            for (int e = 0; e < 8; ++e) {
                bK[e] = kfrag[ct][0][e]; bK[8 + e] = kfrag[ct][1][e];
                bV[e] = vfrag[ct][0][e]; bV[8 + e] = vfrag[ct][1][e];
            }
            accK[ct] = __builtin_amdgcn_wmma_f32_16x16x32_bf16(
                false, a, false, bK, (short)0, accK[ct], false, false);
            accV[ct] = __builtin_amdgcn_wmma_f32_16x16x32_bf16(
                false, a, false, bV, (short)0, accV[ct], false, false);
        }
    }

    // ---- Phase E: normalize, alignment = sigmoid(q.fk/8), write out ----
    // C/D layout: lane holds col n = lane&15; VGPR p holds row p + 8*(lane>=16).
    #pragma unroll
    for (int p = 0; p < 8; ++p) {
        const int r   = Rt * 16 + p + 8 * hiK;       // local out row 0..127
        const int gt  = t0 + r;                      // global seq position
        const float inv = s_invW[r];
        const long rowbase = baseBH + (long)gt * HID;

        float fk[4];
        float dot = 0.f;
        #pragma unroll
        for (int ct = 0; ct < 4; ++ct) {
            fk[ct] = accK[ct][p] * inv;              // focused_k element
            const float qv = q[rowbase + ct * 16 + nlane];
            dot += fk[ct] * qv;
        }
        // reduce across the 16 lanes that share this row (stay in half-wave)
        #pragma unroll
        for (int m = 8; m >= 1; m >>= 1) dot += __shfl_xor(dot, m, 32);

        const float align = 1.f / (1.f + __expf(-dot * SCALE_F));
        #pragma unroll
        for (int ct = 0; ct < 4; ++ct) {
            const float ov = accV[ct][p] * inv * align;   // align * focused_v
            out[rowbase + ct * 16 + nlane] = ov;
        }
    }
}

extern "C" void kernel_launch(void* const* d_in, const int* in_sizes, int n_in,
                              void* d_out, int out_size, void* d_ws, size_t ws_size,
                              hipStream_t stream) {
    const float* q = (const float*)d_in[0];
    const float* f = (const float*)d_in[1];
    const float* k = (const float*)d_in[2];
    const float* v = (const float*)d_in[3];
    float* out = (float*)d_out;

    const int B = in_sizes[0] / (SEQ * HID);     // 4 for the reference shapes
    dim3 grid(SEQ / TILE, NHEAD, B);             // 64 x 16 x 4 = 4096 independent WGs
    leap_wmma_kernel<<<grid, 256, 0, stream>>>(q, f, k, v, out);
}
